// GPT_85624468013746
// MI455X (gfx1250) — matrix operand order repaired and checked by reading, not compiled, not run on hardware
//
#include <hip/hip_runtime.h>
#include <cstdint>
#include <cstddef>

// ---------------- problem constants ----------------
constexpr int CB  = 2;
constexpr int CT  = 2048;
constexpr int CV  = 50257;
constexpr int CD  = 1024;
constexpr int CH  = 16;
constexpr int CFF = 4096;
constexpr int CL  = 8;
constexpr int CHD = 64;           // head dim
constexpr int CM  = CB * CT;      // 4096 rows everywhere

// ---------------- WMMA types ----------------
typedef __bf16 v16bf __attribute__((ext_vector_type(16)));
typedef float  v8f   __attribute__((ext_vector_type(8)));

union FragBF {            // 16 bf16 = 8 VGPRs, loaded as 2x b128
    v16bf v;
    uint4 q[2];
};

static __device__ __forceinline__ v8f wmma_bf16(const FragBF& a, const FragBF& b, v8f c) {
    return __builtin_amdgcn_wmma_f32_16x16x32_bf16(false, a.v, false, b.v,
                                                   (short)0, c, false, false);
}

// ---------------- bf16 helpers (RNE) ----------------
static __device__ __forceinline__ unsigned short f2bf(float f) {
    unsigned int u = __float_as_uint(f);
    u += 0x7FFFu + ((u >> 16) & 1u);
    return (unsigned short)(u >> 16);
}
static __device__ __forceinline__ unsigned int f2bf2(float lo, float hi) {
    return (unsigned)f2bf(lo) | ((unsigned)f2bf(hi) << 16);
}

static __device__ __forceinline__ float gelu_erf(float x) {
    // exact erf GELU, matches jax.nn.gelu(approximate=False)
    return 0.5f * x * (1.0f + erff(x * 0.70710678118654752440f));
}

// =====================================================================
// Embedding: x[b,t,:] = tok_emb[idx[b,t],:] + pos_emb[t,:]
// =====================================================================
__global__ __launch_bounds__(256) void embed_kernel(
    const int* __restrict__ idx, const float* __restrict__ tok,
    const float* __restrict__ pos, float* __restrict__ x)
{
    size_t i = (size_t)blockIdx.x * 256 + threadIdx.x;
    size_t total = (size_t)CM * CD;
    if (i >= total) return;
    int d  = (int)(i & (CD - 1));
    int bt = (int)(i >> 10);          // /CD
    int t  = bt & (CT - 1);
    x[i] = tok[(size_t)idx[bt] * CD + d] + pos[(size_t)t * CD + d];
}

// =====================================================================
// LayerNorm (fp32 in -> bf16 out), one block per row, D = 1024
// =====================================================================
__global__ __launch_bounds__(256) void layernorm_bf16_kernel(
    const float* __restrict__ x, const float* __restrict__ w,
    const float* __restrict__ b, unsigned short* __restrict__ h)
{
    __shared__ float sred[16];
    const int tid  = threadIdx.x;
    const int lane = tid & 31, wid = tid >> 5;
    const size_t base = (size_t)blockIdx.x * CD;

    float v[4], s = 0.f, sq = 0.f;
#pragma unroll
    for (int j = 0; j < 4; ++j) {
        v[j] = x[base + tid + 256 * j];
        s += v[j]; sq += v[j] * v[j];
    }
#pragma unroll
    for (int m = 16; m; m >>= 1) { s += __shfl_xor(s, m, 32); sq += __shfl_xor(sq, m, 32); }
    if (lane == 0) { sred[wid] = s; sred[8 + wid] = sq; }
    __syncthreads();
    float ts = 0.f, tq = 0.f;
#pragma unroll
    for (int i = 0; i < 8; ++i) { ts += sred[i]; tq += sred[8 + i]; }
    const float mean = ts * (1.0f / CD);
    const float var  = tq * (1.0f / CD) - mean * mean;
    const float rstd = rsqrtf(var + 1e-5f);
#pragma unroll
    for (int j = 0; j < 4; ++j) {
        int c = tid + 256 * j;
        h[base + c] = f2bf((v[j] - mean) * rstd * w[c] + b[c]);
    }
}

// =====================================================================
// Tiled GEMM: C[M,N] = act(A[M,K](bf16) * W[N,K]^T(fp32->bf16) + bias) (+resid)
// Block tile 128x128, BK=64, 8 waves (4x2), wave tile 32x64 (2x4 WMMA tiles).
// Double-buffered LDS: one barrier per 64-k iteration; next iteration's
// global loads issued before the 16-WMMA burst. All fragments for a
// k-step are loaded into distinct registers before the WMMA burst so the
// DS waits stage down instead of serializing each WMMA on dscnt==0.
// =====================================================================
template<bool HAS_BIAS, bool DO_GELU, bool HAS_RESID, bool OUT_F, bool OUT_B>
__global__ __launch_bounds__(256) void gemm_bf16_wmma(
    const unsigned short* __restrict__ A, const float* __restrict__ W,
    const float* __restrict__ bias, const float* __restrict__ resid,
    float* __restrict__ outF, unsigned short* __restrict__ outB,
    int M, int N, int K)
{
    constexpr int SA = 72, SB = 72;
    __shared__ unsigned short Alds[2][128 * SA];
    __shared__ unsigned short Blds[2][128 * SB];

    const int tid  = threadIdx.x;
    const int lane = tid & 31;
    const int wid  = tid >> 5;
    const int wm   = wid >> 1;              // 0..3 -> 32-row strip
    const int wn   = wid & 1;               // 0..1 -> 64-col strip
    const int l15  = lane & 15;
    const int khA  = (lane < 16) ? 0 : 8;   // A frag k split
    const int khB  = (lane < 16) ? 0 : 8;   // B frag k split
    const int mblock = blockIdx.y * 128;
    const int nblock = blockIdx.x * 128;

    v8f c[2][4];
#pragma unroll
    for (int i = 0; i < 2; ++i)
#pragma unroll
        for (int j = 0; j < 4; ++j)
#pragma unroll
            for (int r = 0; r < 8; ++r) c[i][j][r] = 0.0f;

    // staging assignments: each thread owns one A row-half and one B row-half
    const int am = tid >> 1, ak = (tid & 1) * 32;   // A: 32 bf16 = 4x b128
    const int bn = tid >> 1, bk = (tid & 1) * 32;   // B: 32 fp32 -> 32 bf16
    const int gn = nblock + bn;
    const unsigned short* agp = A + (size_t)(mblock + am) * K + ak;
    const float*          wgp = (gn < N) ? (W + (size_t)gn * K + bk) : nullptr;

    uint4  ra[4];
    float4 rw[8];

    auto loadStage = [&](int kk) {
#pragma unroll
        for (int j = 0; j < 4; ++j)
            ra[j] = *(const uint4*)(agp + kk + 8 * j);
        if (wgp) {
#pragma unroll
            for (int j = 0; j < 8; ++j)
                rw[j] = *(const float4*)(wgp + kk + 4 * j);
            __builtin_prefetch(wgp + kk + 64, 0, 0);   // global_prefetch_b8
        } else {
#pragma unroll
            for (int j = 0; j < 8; ++j) rw[j] = make_float4(0.f, 0.f, 0.f, 0.f);
        }
    };
    auto storeStage = [&](int buf) {
#pragma unroll
        for (int j = 0; j < 4; ++j)
            *(uint4*)(&Alds[buf][am * SA + ak + 8 * j]) = ra[j];
#pragma unroll
        for (int j = 0; j < 4; ++j) {
            uint4 bw;
            bw.x = f2bf2(rw[2 * j].x, rw[2 * j].y);
            bw.y = f2bf2(rw[2 * j].z, rw[2 * j].w);
            bw.z = f2bf2(rw[2 * j + 1].x, rw[2 * j + 1].y);
            bw.w = f2bf2(rw[2 * j + 1].z, rw[2 * j + 1].w);
            *(uint4*)(&Blds[buf][bn * SB + bk + 8 * j]) = bw;
        }
    };

    const int nIter = K >> 6;          // K / 64
    loadStage(0);
    storeStage(0);

    for (int i = 0; i < nIter; ++i) {
        __syncthreads();
        const int cur = i & 1;
        const bool more = (i + 1) < nIter;
        if (more) loadStage((i + 1) << 6);       // hide under WMMA burst

#pragma unroll
        for (int s = 0; s < 2; ++s) {
            // batch ALL fragment loads for this k-step before any WMMA
            FragBF a[2], b[4];
#pragma unroll
            for (int tm = 0; tm < 2; ++tm) {
                const int row = wm * 32 + tm * 16 + l15;
                a[tm].q[0] = *(const uint4*)(&Alds[cur][row * SA + s * 32 + khA]);
                a[tm].q[1] = *(const uint4*)(&Alds[cur][row * SA + s * 32 + khA + 16]);
            }
#pragma unroll
            for (int tn = 0; tn < 4; ++tn) {
                const int nn = wn * 64 + tn * 16 + l15;
                b[tn].q[0] = *(const uint4*)(&Blds[cur][nn * SB + s * 32 + khB]);
                b[tn].q[1] = *(const uint4*)(&Blds[cur][nn * SB + s * 32 + khB + 16]);
            }
#pragma unroll
            for (int tn = 0; tn < 4; ++tn) {
                c[0][tn] = wmma_bf16(a[0], b[tn], c[0][tn]);
                c[1][tn] = wmma_bf16(a[1], b[tn], c[1][tn]);
            }
        }
        if (more) storeStage(cur ^ 1);
    }

    // ---- epilogue ----
#pragma unroll
    for (int tm = 0; tm < 2; ++tm) {
#pragma unroll
        for (int tn = 0; tn < 4; ++tn) {
            const int colb = nblock + wn * 64 + tn * 16 + l15;
            if (colb >= N) continue;
            const float bv = HAS_BIAS ? bias[colb] : 0.0f;
#pragma unroll
            for (int r = 0; r < 8; ++r) {
                const int row = mblock + wm * 32 + tm * 16 + ((lane < 16) ? r : 8 + r);
                float v = c[tm][tn][r] + bv;
                if (DO_GELU)  v = gelu_erf(v);
                if (HAS_RESID) v += resid[(size_t)row * N + colb];
                if (OUT_F) outF[(size_t)row * N + colb] = v;
                if (OUT_B) outB[(size_t)row * N + colb] = f2bf(v);
            }
        }
    }
}

// =====================================================================
// Flash attention: 4 waves per workgroup share one K/V staging; each
// wave owns a 16-query strip (64 queries per block -> 4x less K/V
// traffic). S = Q K^T and O += P V via WMMA; per-wave online softmax
// through a private LDS slice (one shfl_xor(16) combine per row).
// All B fragments of each WMMA phase are loaded before the burst.
// =====================================================================
__global__ __launch_bounds__(128) void attn_kernel(
    const unsigned short* __restrict__ qkv,   // [B*T][3D] bf16
    unsigned short* __restrict__ y)           // [B*T][D]  bf16
{
    constexpr int SK = 72;   // Klds [key=32][hd=64]
    constexpr int SV = 40;   // Vlds [hd=64][key=32]
    constexpr int SS = 36;   // Slds [q=16][key=32] f32   (per wave)
    constexpr int SP = 40;   // Plds [q=16][key=32] bf16  (per wave)
    __shared__ unsigned short Klds[32 * SK];
    __shared__ unsigned short Vlds[64 * SV];
    __shared__ float          Slds[4][16 * SS];
    __shared__ unsigned short Plds[4][16 * SP];
    __shared__ float          stat[4][16];     // alpha / invL broadcast

    const int tid  = threadIdx.x;
    const int lane = tid & 31;
    const int wid  = tid >> 5;                 // wave -> query strip
    const int l15  = lane & 15;
    const int half = lane >> 4;
    const int q0g  = blockIdx.x * 64;          // workgroup query base
    const int q0   = q0g + wid * 16;           // this wave's strip
    const int h    = blockIdx.y;
    const int b    = blockIdx.z;
    const int bt0  = b * CT + q0;
    const int qoff = h * CHD;
    const int koff = CD + h * CHD;
    const int voff = 2 * CD + h * CHD;

    float* mySl          = Slds[wid];
    unsigned short* myPl = Plds[wid];
    float* myStat        = stat[wid];

    // ---- resident Q A-fragments (16 q x 64 hd) ----
    FragBF qa[2];
    {
        const int kA = (lane < 16) ? 0 : 8;
        const unsigned short* qrow = qkv + (size_t)(bt0 + l15) * (3 * CD) + qoff;
#pragma unroll
        for (int f = 0; f < 2; ++f) {
            qa[f].q[0] = *(const uint4*)(qrow + f * 32 + kA);
            qa[f].q[1] = *(const uint4*)(qrow + f * 32 + kA + 16);
        }
    }

    // per-lane row stats: this lane tracks query row (l15) of its strip
    float Mr = -1e30f, Lr = 0.0f;
    v8f o[4];
#pragma unroll
    for (int t = 0; t < 4; ++t)
#pragma unroll
        for (int r = 0; r < 8; ++r) o[t][r] = 0.0f;

    // staging assignment: thread -> (key, 16-hd segment)
    const int skey = tid >> 2;          // 0..31
    const int sseg = tid & 3;           // 0..3

    const float scale = 0.125f;         // 1/sqrt(64)
    const int kbmax = (q0g + 63) >> 5;  // workgroup-wide causal bound
    for (int kb = 0; kb <= kbmax; ++kb) {
        const int kbase = kb * 32;
        const bool active = (kbase <= q0 + 15);

        // ---- cooperative K/V staging: batch all global loads first ----
        {
            const size_t rowb = (size_t)(b * CT + kbase + skey) * (3 * CD);
            const unsigned short* kr = qkv + rowb + koff + sseg * 16;
            const unsigned short* vr = qkv + rowb + voff + sseg * 16;
            uint4 k0 = *(const uint4*)(kr);
            uint4 k1 = *(const uint4*)(kr + 8);
            uint4 v0 = *(const uint4*)(vr);
            uint4 v1 = *(const uint4*)(vr + 8);
            // K: straight copy [key][hd]  (this IS the B^T layout for S)
            *(uint4*)(&Klds[skey * SK + sseg * 16])     = k0;
            *(uint4*)(&Klds[skey * SK + sseg * 16 + 8]) = k1;
            // V: transposed scatter -> [hd][key]
            union { uint4 q; unsigned short e[8]; } u0, u1;
            u0.q = v0; u1.q = v1;
#pragma unroll
            for (int e = 0; e < 8; ++e) {
                Vlds[(sseg * 16 + e) * SV + skey]     = u0.e[e];
                Vlds[(sseg * 16 + 8 + e) * SV + skey] = u1.e[e];
            }
        }
        __syncthreads();

        // ---- S = Q K^T : 2 key n-tiles, 2 k-steps over hd ----
        if (active) {
            // batch ALL 4 K-fragments, then the 4-WMMA burst
            FragBF bk[2][2];     // [tile][kstep]
            const int kh = (lane < 16) ? 0 : 8;
#pragma unroll
            for (int t = 0; t < 2; ++t) {
                const int nn = t * 16 + l15;
                bk[t][0].q[0] = *(const uint4*)(&Klds[nn * SK + kh]);
                bk[t][0].q[1] = *(const uint4*)(&Klds[nn * SK + kh + 16]);
                bk[t][1].q[0] = *(const uint4*)(&Klds[nn * SK + 32 + kh]);
                bk[t][1].q[1] = *(const uint4*)(&Klds[nn * SK + 32 + kh + 16]);
            }
            v8f s[2];
#pragma unroll
            for (int t = 0; t < 2; ++t) {
#pragma unroll
                for (int r = 0; r < 8; ++r) s[t][r] = 0.0f;
                s[t] = wmma_bf16(qa[0], bk[t][0], s[t]);
                s[t] = wmma_bf16(qa[1], bk[t][1], s[t]);
            }
            // dump masked, scaled S (C-layout) into this wave's LDS tile
#pragma unroll
            for (int t = 0; t < 2; ++t)
#pragma unroll
                for (int r = 0; r < 8; ++r) {
                    const int row = (lane < 16) ? r : 8 + r;
                    const int gq  = q0 + row;
                    const int gk  = kbase + t * 16 + l15;
                    mySl[row * SS + t * 16 + l15] = (gk <= gq) ? s[t][r] * scale : -1e30f;
                }
        }
        __syncthreads();

        // ---- per-row softmax: lane pair (l15, half) owns row l15 ----
        if (active) {
            const float* sr = &mySl[l15 * SS + half * 16];
            float4 c0 = *(const float4*)(sr);
            float4 c1 = *(const float4*)(sr + 4);
            float4 c2 = *(const float4*)(sr + 8);
            float4 c3 = *(const float4*)(sr + 12);
            float mx = fmaxf(fmaxf(fmaxf(c0.x, c0.y), fmaxf(c0.z, c0.w)),
                             fmaxf(fmaxf(c1.x, c1.y), fmaxf(c1.z, c1.w)));
            mx = fmaxf(mx, fmaxf(fmaxf(fmaxf(c2.x, c2.y), fmaxf(c2.z, c2.w)),
                                 fmaxf(fmaxf(c3.x, c3.y), fmaxf(c3.z, c3.w))));
            mx = fmaxf(mx, __shfl_xor(mx, 16, 32));
            const float Mn = fmaxf(Mr, mx);
            const float al = __expf(Mr - Mn);
            Mr = Mn;
            float e[16];
            e[0]  = __expf(c0.x - Mn); e[1]  = __expf(c0.y - Mn);
            e[2]  = __expf(c0.z - Mn); e[3]  = __expf(c0.w - Mn);
            e[4]  = __expf(c1.x - Mn); e[5]  = __expf(c1.y - Mn);
            e[6]  = __expf(c1.z - Mn); e[7]  = __expf(c1.w - Mn);
            e[8]  = __expf(c2.x - Mn); e[9]  = __expf(c2.y - Mn);
            e[10] = __expf(c2.z - Mn); e[11] = __expf(c2.w - Mn);
            e[12] = __expf(c3.x - Mn); e[13] = __expf(c3.y - Mn);
            e[14] = __expf(c3.z - Mn); e[15] = __expf(c3.w - Mn);
            float sum = 0.0f;
#pragma unroll
            for (int i = 0; i < 16; ++i) sum += e[i];
            sum += __shfl_xor(sum, 16, 32);
            Lr = Lr * al + sum;
            if (lane < 16) myStat[l15] = al;
            uint4 p0, p1;
            p0.x = f2bf2(e[0],  e[1]);  p0.y = f2bf2(e[2],  e[3]);
            p0.z = f2bf2(e[4],  e[5]);  p0.w = f2bf2(e[6],  e[7]);
            p1.x = f2bf2(e[8],  e[9]);  p1.y = f2bf2(e[10], e[11]);
            p1.z = f2bf2(e[12], e[13]); p1.w = f2bf2(e[14], e[15]);
            *(uint4*)(&myPl[l15 * SP + half * 16])     = p0;
            *(uint4*)(&myPl[l15 * SP + half * 16 + 8]) = p1;
        }
        __syncthreads();

        // ---- rescale O, then O += P V : 4 hd n-tiles, k = 32 keys ----
        if (active) {
#pragma unroll
            for (int r = 0; r < 8; ++r) {
                const float al = myStat[(lane < 16) ? r : 8 + r];
#pragma unroll
                for (int t = 0; t < 4; ++t) o[t][r] *= al;
            }
            // batch P-fragment and ALL 4 V-fragments, then the WMMA burst
            FragBF pa, bv[4];
            const int pk = (lane < 16) ? 0 : 8;
            pa.q[0] = *(const uint4*)(&myPl[l15 * SP + pk]);
            pa.q[1] = *(const uint4*)(&myPl[l15 * SP + pk + 16]);
            const int kh = (lane < 16) ? 0 : 8;
#pragma unroll
            for (int t = 0; t < 4; ++t) {
                const int nn = t * 16 + l15;
                bv[t].q[0] = *(const uint4*)(&Vlds[nn * SV + kh]);
                bv[t].q[1] = *(const uint4*)(&Vlds[nn * SV + kh + 16]);
            }
#pragma unroll
            for (int t = 0; t < 4; ++t)
                o[t] = wmma_bf16(pa, bv[t], o[t]);
        }
        __syncthreads();
    }

    // ---- normalize and store y (bf16) ----
    if (lane < 16) myStat[l15] = 1.0f / Lr;
    __syncthreads();
#pragma unroll
    for (int t = 0; t < 4; ++t)
#pragma unroll
        for (int r = 0; r < 8; ++r) {
            const int rr  = (lane < 16) ? r : 8 + r;
            const int row = bt0 + rr;
            const int col = h * CHD + t * 16 + l15;
            y[(size_t)row * CD + col] = f2bf(o[t][r] * myStat[rr]);
        }
}

// =====================================================================
// log-softmax NLL per row (deterministic two-stage mean)
// =====================================================================
__global__ __launch_bounds__(256) void loss_rows_kernel(
    const float* __restrict__ logits, const int* __restrict__ tgt,
    float* __restrict__ nll)
{
    __shared__ float sred[8];
    const int tid = threadIdx.x, lane = tid & 31, wid = tid >> 5;
    const float* lr = logits + (size_t)blockIdx.x * CV;

    float m = -1e30f;
    for (int i = tid; i < CV; i += 256) m = fmaxf(m, lr[i]);
#pragma unroll
    for (int s = 16; s; s >>= 1) m = fmaxf(m, __shfl_xor(m, s, 32));
    if (lane == 0) sred[wid] = m;
    __syncthreads();
    float mx = sred[0];
#pragma unroll
    for (int i = 1; i < 8; ++i) mx = fmaxf(mx, sred[i]);
    __syncthreads();

    float sum = 0.0f;
    for (int i = tid; i < CV; i += 256) sum += __expf(lr[i] - mx);
#pragma unroll
    for (int s = 16; s; s >>= 1) sum += __shfl_xor(sum, s, 32);
    if (lane == 0) sred[wid] = sum;
    __syncthreads();
    if (tid == 0) {
        float tot = 0.0f;
#pragma unroll
        for (int i = 0; i < 8; ++i) tot += sred[i];
        const int t = tgt[blockIdx.x];
        nll[blockIdx.x] = (logf(tot) + mx) - lr[t];
    }
}

__global__ __launch_bounds__(256) void loss_reduce_kernel(
    const float* __restrict__ nll, float* __restrict__ loss_out)
{
    __shared__ float sred[8];
    const int tid = threadIdx.x, lane = tid & 31, wid = tid >> 5;
    float s = 0.0f;
    for (int i = tid; i < CM; i += 256) s += nll[i];
#pragma unroll
    for (int m = 16; m; m >>= 1) s += __shfl_xor(s, m, 32);
    if (lane == 0) sred[wid] = s;
    __syncthreads();
    if (tid == 0) {
        float tot = 0.0f;
#pragma unroll
        for (int i = 0; i < 8; ++i) tot += sred[i];
        *loss_out = tot * (1.0f / (float)CM);
    }
}

// =====================================================================
// host-side orchestration
// =====================================================================
extern "C" void kernel_launch(void* const* d_in, const int* in_sizes, int n_in,
                              void* d_out, int out_size, void* d_ws, size_t ws_size,
                              hipStream_t stream)
{
    (void)in_sizes; (void)n_in; (void)out_size; (void)ws_size;

    const int*   idx     = (const int*)  d_in[0];
    const int*   targets = (const int*)  d_in[1];
    const float* tok_emb = (const float*)d_in[2];
    const float* pos_emb = (const float*)d_in[3];
    const float* ln1_w   = (const float*)d_in[4];
    const float* ln1_b   = (const float*)d_in[5];
    const float* qkv_w   = (const float*)d_in[6];
    const float* qkv_b   = (const float*)d_in[7];
    const float* proj_w  = (const float*)d_in[8];
    const float* proj_b  = (const float*)d_in[9];
    const float* ln2_w   = (const float*)d_in[10];
    const float* ln2_b   = (const float*)d_in[11];
    const float* fc1_w   = (const float*)d_in[12];
    const float* fc1_b   = (const float*)d_in[13];
    const float* fc2_w   = (const float*)d_in[14];
    const float* fc2_b   = (const float*)d_in[15];
    const float* lnf_w   = (const float*)d_in[16];
    const float* lnf_b   = (const float*)d_in[17];
    const float* lm_w    = (const float*)d_in[18];

    float* logits = (float*)d_out;
    float* loss   = logits + (size_t)CM * CV;

    // workspace layout
    char* w = (char*)d_ws;
    float*          x    = (float*)w;           w += (size_t)CM * CD * 4;      // 16.8 MB
    unsigned short* hbuf = (unsigned short*)w;  w += (size_t)CM * CD * 2;      // 8.4 MB
    unsigned short* qkvb = (unsigned short*)w;  w += (size_t)CM * 3 * CD * 2;  // 25.2 MB
    unsigned short* ybuf = (unsigned short*)w;  w += (size_t)CM * CD * 2;      // 8.4 MB
    unsigned short* m1   = (unsigned short*)w;  w += (size_t)CM * CFF * 2;     // 33.6 MB
    float*          nll  = (float*)w;           w += (size_t)CM * 4;           // 16 KB

    const dim3 blk256(256), blk128(128);

    embed_kernel<<<(unsigned)(((size_t)CM * CD + 255) / 256), blk256, 0, stream>>>(
        idx, tok_emb, pos_emb, x);

    const dim3 gQKV((3 * CD + 127) / 128, CM / 128);
    const dim3 gPRJ((CD + 127) / 128,     CM / 128);
    const dim3 gFC1((CFF + 127) / 128,    CM / 128);
    const dim3 gFC2((CD + 127) / 128,     CM / 128);
    const dim3 gLMH((CV + 127) / 128,     CM / 128);
    const dim3 gATT(CT / 64, CH, CB);

    for (int l = 0; l < CL; ++l) {
        layernorm_bf16_kernel<<<CM, blk256, 0, stream>>>(
            x, ln1_w + (size_t)l * CD, ln1_b + (size_t)l * CD, hbuf);
        gemm_bf16_wmma<true, false, false, false, true><<<gQKV, blk256, 0, stream>>>(
            hbuf, qkv_w + (size_t)l * 3 * CD * CD, qkv_b + (size_t)l * 3 * CD,
            nullptr, nullptr, qkvb, CM, 3 * CD, CD);
        attn_kernel<<<gATT, blk128, 0, stream>>>(qkvb, ybuf);
        gemm_bf16_wmma<true, false, true, true, false><<<gPRJ, blk256, 0, stream>>>(
            ybuf, proj_w + (size_t)l * CD * CD, proj_b + (size_t)l * CD,
            x, x, nullptr, CM, CD, CD);
        layernorm_bf16_kernel<<<CM, blk256, 0, stream>>>(
            x, ln2_w + (size_t)l * CD, ln2_b + (size_t)l * CD, hbuf);
        gemm_bf16_wmma<true, true, false, false, true><<<gFC1, blk256, 0, stream>>>(
            hbuf, fc1_w + (size_t)l * CFF * CD, fc1_b + (size_t)l * CFF,
            nullptr, nullptr, m1, CM, CFF, CD);
        gemm_bf16_wmma<true, false, true, true, false><<<gFC2, blk256, 0, stream>>>(
            m1, fc2_w + (size_t)l * CD * CFF, fc2_b + (size_t)l * CD,
            x, x, nullptr, CM, CD, CFF);
    }

    layernorm_bf16_kernel<<<CM, blk256, 0, stream>>>(x, lnf_w, lnf_b, hbuf);
    gemm_bf16_wmma<false, false, false, true, false><<<gLMH, blk256, 0, stream>>>(
        hbuf, lm_w, nullptr, nullptr, logits, nullptr, CM, CV, CD);

    loss_rows_kernel<<<CM, blk256, 0, stream>>>(logits, targets, nll);
    loss_reduce_kernel<<<1, blk256, 0, stream>>>(nll, loss);
}
